// Iterative_Model_57707180589817
// MI455X (gfx1250) — compile-verified, matrix operand order checked by reading, not run on hardware
//
#include <hip/hip_runtime.h>

typedef __attribute__((ext_vector_type(16))) __bf16 v16bf;
typedef __attribute__((ext_vector_type(8)))  __bf16 v8bf;
typedef __attribute__((ext_vector_type(4)))  __bf16 v4bf;
typedef __attribute__((ext_vector_type(8)))  float  v8f;

#define B_ROWS 8192
#define K_DIM  4096
#define N_DIM  2048
#define NCLS   5
#define NSAMP  8
#define NPROTO (NCLS * NSAMP)

// Concatenate two 8-wide bf16 vectors into the 16-wide WMMA fragment.
#define CAT16(a, b) __builtin_shufflevector((a), (b), 0, 1, 2, 3, 4, 5, 6, 7, \
                                            8, 9, 10, 11, 12, 13, 14, 15)

// Round-to-nearest-even f32 -> bf16, plus residual term for hi/lo split.
__device__ __forceinline__ void bf16_split(float f, __bf16& hi, __bf16& lo) {
    unsigned u  = __builtin_bit_cast(unsigned, f);
    unsigned r  = (u + 0x7FFFu + ((u >> 16) & 1u)) >> 16;
    float    fh = __builtin_bit_cast(float, r << 16);
    float    rem = f - fh;
    unsigned u2 = __builtin_bit_cast(unsigned, rem);
    unsigned r2 = (u2 + 0x7FFFu + ((u2 >> 16) & 1u)) >> 16;
    hi = __builtin_bit_cast(__bf16, (unsigned short)r);
    lo = __builtin_bit_cast(__bf16, (unsigned short)r2);
}

// --- Kernel 1a: x [M][K] f32 -> Xh/Xl bf16 (same layout, hi/lo split) ---
__global__ __launch_bounds__(256)
void convert_x(const float* __restrict__ x,
               __bf16* __restrict__ Xh, __bf16* __restrict__ Xl) {
    const size_t i = ((size_t)blockIdx.x * 256 + threadIdx.x) * 4;
    float4 v = *(const float4*)(x + i);
    v4bf h, l;
    __bf16 hh, ll;
    bf16_split(v.x, hh, ll); h[0] = hh; l[0] = ll;
    bf16_split(v.y, hh, ll); h[1] = hh; l[1] = ll;
    bf16_split(v.z, hh, ll); h[2] = hh; l[2] = ll;
    bf16_split(v.w, hh, ll); h[3] = hh; l[3] = ll;
    *(v4bf*)(Xh + i) = h;
    *(v4bf*)(Xl + i) = l;
}

// --- Kernel 1b: W_feat [K][N] f32 -> Wh/Wl [N][K] bf16 (transposed, hi/lo) ---
__global__ __launch_bounds__(256)
void convert_transpose_w(const float* __restrict__ W,
                         __bf16* __restrict__ Wh, __bf16* __restrict__ Wl) {
    __shared__ float tile[32][33];
    const int tx = threadIdx.x;          // 0..31
    const int ty = threadIdx.y;          // 0..7
    const int nb = blockIdx.x * 32;
    const int kb = blockIdx.y * 32;
    #pragma unroll
    for (int j = 0; j < 4; ++j) {
        int kl = ty + j * 8;
        tile[kl][tx] = W[(size_t)(kb + kl) * N_DIM + nb + tx];
    }
    __syncthreads();
    #pragma unroll
    for (int j = 0; j < 4; ++j) {
        int nl = ty + j * 8;
        float v = tile[tx][nl];
        __bf16 hi, lo;
        bf16_split(v, hi, lo);
        size_t o = (size_t)(nb + nl) * K_DIM + (kb + tx);
        Wh[o] = hi;
        Wl[o] = lo;
    }
}

// --- Kernel 2: normalize the 40 prototype rows ---
__global__ __launch_bounds__(256)
void proto_normalize(const float* __restrict__ P, float* __restrict__ Pn) {
    const int r   = blockIdx.x;
    const int tid = threadIdx.x;
    __shared__ float red[256];
    float s = 0.f;
    for (int d = tid; d < N_DIM; d += 256) {
        float v = P[(size_t)r * N_DIM + d];
        s += v * v;
    }
    red[tid] = s;
    __syncthreads();
    for (int o = 128; o > 0; o >>= 1) {
        if (tid < o) red[tid] += red[tid + o];
        __syncthreads();
    }
    float inv = rsqrtf(red[0]);
    for (int d = tid; d < N_DIM; d += 256) {
        Pn[(size_t)r * N_DIM + d] = P[(size_t)r * N_DIM + d] * inv;
    }
}

// --- Kernel 3: feats = x @ W_feat, bf16x3 WMMA (hi*hi + hi*lo + lo*hi) ---
// 8 waves/block (2 M x 4 N); each wave owns a 32x32 block = 4 accumulators.
// Inner loop: 16x 16-byte loads, 12 WMMAs, no conversion VALU.
__global__ __launch_bounds__(256)
void feat_gemm(const __bf16* __restrict__ Xh, const __bf16* __restrict__ Xl,
               const __bf16* __restrict__ Wh, const __bf16* __restrict__ Wl,
               float* __restrict__ feats) {
    const int lane  = threadIdx.x & 31;
    const int wave  = threadIdx.x >> 5;
    const int waveN = wave & 3;
    const int waveM = wave >> 2;
    const int nTile = blockIdx.x * 128 + waveN * 32;
    const int mTile = blockIdx.y * 64 + waveM * 32;
    const int row   = lane & 15;   // A: M index / B: N index within tile
    const int half  = lane >> 4;   // K-chunk interleave: {0..7,16..23} vs {8..15,24..31}

    const __bf16* xh = Xh + (size_t)(mTile + row) * K_DIM + half * 8;
    const __bf16* xl = Xl + (size_t)(mTile + row) * K_DIM + half * 8;
    const __bf16* wh = Wh + (size_t)(nTile + row) * K_DIM + half * 8;
    const __bf16* wl = Wl + (size_t)(nTile + row) * K_DIM + half * 8;
    const size_t  T  = (size_t)16 * K_DIM;   // next 16-row tile, same array

    v8f acc00 = {}, acc01 = {}, acc10 = {}, acc11 = {};

    for (int k = 0; k < K_DIM; k += 32) {
        v16bf A0h = CAT16(*(const v8bf*)(xh + k),     *(const v8bf*)(xh + k + 16));
        v16bf A0l = CAT16(*(const v8bf*)(xl + k),     *(const v8bf*)(xl + k + 16));
        v16bf A1h = CAT16(*(const v8bf*)(xh + T + k), *(const v8bf*)(xh + T + k + 16));
        v16bf A1l = CAT16(*(const v8bf*)(xl + T + k), *(const v8bf*)(xl + T + k + 16));
        v16bf B0h = CAT16(*(const v8bf*)(wh + k),     *(const v8bf*)(wh + k + 16));
        v16bf B0l = CAT16(*(const v8bf*)(wl + k),     *(const v8bf*)(wl + k + 16));
        v16bf B1h = CAT16(*(const v8bf*)(wh + T + k), *(const v8bf*)(wh + T + k + 16));
        v16bf B1l = CAT16(*(const v8bf*)(wl + T + k), *(const v8bf*)(wl + T + k + 16));

        acc00 = __builtin_amdgcn_wmma_f32_16x16x32_bf16(false, A0h, false, B0h,
                                                        (short)0, acc00, false, false);
        acc01 = __builtin_amdgcn_wmma_f32_16x16x32_bf16(false, A0h, false, B1h,
                                                        (short)0, acc01, false, false);
        acc10 = __builtin_amdgcn_wmma_f32_16x16x32_bf16(false, A1h, false, B0h,
                                                        (short)0, acc10, false, false);
        acc11 = __builtin_amdgcn_wmma_f32_16x16x32_bf16(false, A1h, false, B1h,
                                                        (short)0, acc11, false, false);

        acc00 = __builtin_amdgcn_wmma_f32_16x16x32_bf16(false, A0h, false, B0l,
                                                        (short)0, acc00, false, false);
        acc01 = __builtin_amdgcn_wmma_f32_16x16x32_bf16(false, A0h, false, B1l,
                                                        (short)0, acc01, false, false);
        acc10 = __builtin_amdgcn_wmma_f32_16x16x32_bf16(false, A1h, false, B0l,
                                                        (short)0, acc10, false, false);
        acc11 = __builtin_amdgcn_wmma_f32_16x16x32_bf16(false, A1h, false, B1l,
                                                        (short)0, acc11, false, false);

        acc00 = __builtin_amdgcn_wmma_f32_16x16x32_bf16(false, A0l, false, B0h,
                                                        (short)0, acc00, false, false);
        acc01 = __builtin_amdgcn_wmma_f32_16x16x32_bf16(false, A0l, false, B1h,
                                                        (short)0, acc01, false, false);
        acc10 = __builtin_amdgcn_wmma_f32_16x16x32_bf16(false, A1l, false, B0h,
                                                        (short)0, acc10, false, false);
        acc11 = __builtin_amdgcn_wmma_f32_16x16x32_bf16(false, A1l, false, B1h,
                                                        (short)0, acc11, false, false);
    }

    // C/D layout: VGPR v, lanes 0-15 -> M=v, lanes 16-31 -> M=8+v; N=lane&15.
    const int cRow = half * 8;
    {
        float* p = feats + (size_t)(mTile + cRow) * N_DIM + nTile + row;
        #pragma unroll
        for (int v = 0; v < 8; ++v) p[(size_t)v * N_DIM] = acc00[v];
    }
    {
        float* p = feats + (size_t)(mTile + cRow) * N_DIM + nTile + 16 + row;
        #pragma unroll
        for (int v = 0; v < 8; ++v) p[(size_t)v * N_DIM] = acc01[v];
    }
    {
        float* p = feats + (size_t)(mTile + 16 + cRow) * N_DIM + nTile + row;
        #pragma unroll
        for (int v = 0; v < 8; ++v) p[(size_t)v * N_DIM] = acc10[v];
    }
    {
        float* p = feats + (size_t)(mTile + 16 + cRow) * N_DIM + nTile + 16 + row;
        #pragma unroll
        for (int v = 0; v < 8; ++v) p[(size_t)v * N_DIM] = acc11[v];
    }
}

// --- Kernel 4: logits + prototype scores + label per row (one wave / row) ---
__global__ __launch_bounds__(256)
void head_kernel(const float* __restrict__ feats, const float* __restrict__ W_fc,
                 const float* __restrict__ b_fc, const float* __restrict__ Pn,
                 const int* __restrict__ epoch, float* __restrict__ out,
                 long long* __restrict__ yhat) {
    const int lane = threadIdx.x & 31;
    const int wave = threadIdx.x >> 5;
    const int b    = blockIdx.x * 8 + wave;
    const float* f = feats + (size_t)b * N_DIM;

    float fc[NCLS]   = {};
    float pd[NPROTO] = {};
    for (int d = lane; d < N_DIM; d += 32) {
        const float fv = f[d];
        #pragma unroll
        for (int c = 0; c < NCLS; ++c)
            fc[c] = fmaf(fv, W_fc[(size_t)d * NCLS + c], fc[c]);
        #pragma unroll
        for (int r = 0; r < NPROTO; ++r)
            pd[r] = fmaf(fv, Pn[(size_t)r * N_DIM + d], pd[r]);
    }
    #pragma unroll
    for (int off = 16; off > 0; off >>= 1) {
        #pragma unroll
        for (int c = 0; c < NCLS; ++c) fc[c] += __shfl_down(fc[c], off, 32);
        #pragma unroll
        for (int r = 0; r < NPROTO; ++r) pd[r] += __shfl_down(pd[r], off, 32);
    }
    if (lane == 0) {
        float logits[NCLS];
        #pragma unroll
        for (int c = 0; c < NCLS; ++c) {
            logits[c] = fc[c] + b_fc[c];
            out[(size_t)b * NCLS + c] = logits[c];
        }
        int bi = 0;
        if (*epoch > 2) {
            // argmax over sum_s dot(f, pn[c,s]); the 1/(8*||f||) scale is a
            // positive per-row constant and does not change the argmax.
            float best = -3.4e38f;
            #pragma unroll
            for (int c = 0; c < NCLS; ++c) {
                float s = 0.f;
                #pragma unroll
                for (int j = 0; j < NSAMP; ++j) s += pd[c * NSAMP + j];
                if (s > best) { best = s; bi = c; }
            }
        } else {
            float best = -3.4e38f;
            #pragma unroll
            for (int c = 0; c < NCLS; ++c)
                if (logits[c] > best) { best = logits[c]; bi = c; }
        }
        yhat[b] = (long long)bi;
    }
}

extern "C" void kernel_launch(void* const* d_in, const int* in_sizes, int n_in,
                              void* d_out, int out_size, void* d_ws, size_t ws_size,
                              hipStream_t stream) {
    (void)in_sizes; (void)n_in; (void)out_size; (void)ws_size;

    const float* x      = (const float*)d_in[0];
    const float* W_feat = (const float*)d_in[1];
    const float* W_fc   = (const float*)d_in[2];
    const float* b_fc   = (const float*)d_in[3];
    const float* proto  = (const float*)d_in[4];
    const int*   epoch  = (const int*)d_in[5];

    // Workspace layout (all offsets 256B aligned).
    char*        ws     = (char*)d_ws;
    const size_t xBytes = (size_t)B_ROWS * K_DIM * sizeof(__bf16);  // 67,108,864
    const size_t wBytes = (size_t)N_DIM * K_DIM * sizeof(__bf16);   // 16,777,216
    __bf16* Xh    = (__bf16*)(ws);
    __bf16* Xl    = (__bf16*)(ws + xBytes);
    __bf16* Wh    = (__bf16*)(ws + 2 * xBytes);
    __bf16* Wl    = (__bf16*)(ws + 2 * xBytes + wBytes);
    float*  Pn    = (float*)(ws + 2 * xBytes + 2 * wBytes);
    float*  feats = (float*)(ws + 2 * xBytes + 2 * wBytes +
                             (size_t)NPROTO * N_DIM * sizeof(float));

    float*     outF = (float*)d_out;                               // [8192 x 5] f32
    long long* yhat = (long long*)(outF + (size_t)B_ROWS * NCLS);  // [8192] i64

    convert_x<<<(B_ROWS * (K_DIM / 4)) / 256, 256, 0, stream>>>(x, Xh, Xl);
    convert_transpose_w<<<dim3(N_DIM / 32, K_DIM / 32), dim3(32, 8), 0, stream>>>(
        W_feat, Wh, Wl);
    proto_normalize<<<NPROTO, 256, 0, stream>>>(proto, Pn);
    feat_gemm<<<dim3(N_DIM / 128, B_ROWS / 64), 256, 0, stream>>>(
        Xh, Xl, Wh, Wl, feats);
    head_kernel<<<B_ROWS / 8, 256, 0, stream>>>(feats, W_fc, b_fc, Pn, epoch,
                                                outF, yhat);
}
